// UWBPoseEncoder_27324581937450
// MI455X (gfx1250) — matrix operand order repaired and checked by reading, not compiled, not run on hardware
//
#include <hip/hip_runtime.h>
#include <hip/hip_bf16.h>

// ---------------------------------------------------------------------------
// UWB pose encoder for MI455X (gfx1250): f16 WMMA for all GEMM-shaped work,
// f32 softmax/statistics, flash-attention over M=2048 per batch.
// K/V are written by the encoder in per-tile fragment-swizzled B layout, so
// the attention staging is a verbatim contiguous copy -> async load-to-LDS.
// ---------------------------------------------------------------------------

typedef __attribute__((ext_vector_type(16))) _Float16 v16h;
typedef __attribute__((ext_vector_type(8)))  _Float16 v8h;
typedef __attribute__((ext_vector_type(8)))  float    v8f;

#define NB   32
#define NM   2048
#define NH   64
#define NT   8
#define QT   128   // query rows per block
#define KVT  64    // kv rows per inner tile
#define TILE_ELEMS (KVT * NH)   // 4096 halves = 8 KB per swizzled KV tile

#if defined(__gfx1250__) && \
    __has_builtin(__builtin_amdgcn_global_load_async_to_lds_b128)
#define HAVE_ASYNC_LDS 1
#endif

__device__ inline void copy16_to_lds(const _Float16* g, _Float16* l) {
#ifdef HAVE_ASYNC_LDS
  // Builtin signature (from hipcc diagnostic): param 0 is
  // '__attribute__((vector_size(16))) int __device__ *' (global AS),
  // param 1 the LDS-side pointer, then imm offset and imm cpol.
  typedef int v4i_gcc __attribute__((vector_size(16)));
  typedef __attribute__((address_space(1))) v4i_gcc* gp_t;
  typedef __attribute__((address_space(3))) v4i_gcc* lp_t;
  __builtin_amdgcn_global_load_async_to_lds_b128((gp_t)g, (lp_t)l, 0, 0);
#else
  *(v8h*)l = *(const v8h*)g;
#endif
}
__device__ inline void async_copy_wait() {
#ifdef HAVE_ASYNC_LDS
#if __has_builtin(__builtin_amdgcn_s_wait_asynccnt)
  __builtin_amdgcn_s_wait_asynccnt(0);
#else
  asm volatile("s_wait_asynccnt 0" ::: "memory");
#endif
#endif
}

// ---- A fragment (16x32 f16) from a row-major [rows][stride] LDS array ------
// Lane L: row = L&15, kb = (L<16)?0:16.  Elements 0..15 = K kb..kb+15
// -> two aligned 16-byte LDS loads.
__device__ inline v16h make_a_frag(const _Float16* s, int lane, int row0,
                                   int koff, int stride) {
  int m  = lane & 15;
  int kb = (lane < 16) ? 0 : 16;
  const _Float16* p = s + (row0 + m) * stride + koff + kb;
  v8h lo = *(const v8h*)(p);
  v8h hi = *(const v8h*)(p + 8);
  return __builtin_shufflevector(lo, hi, 0, 1, 2, 3, 4, 5, 6, 7,
                                 8, 9, 10, 11, 12, 13, 14, 15);
}

// ---- B fragment from fragment-major swizzled storage ----------------------
// Swizzled layout: element (k, n) of a K x 64 B-matrix lives at
//   frag = (k>>5)*4 + (n>>4)
//   addr = (frag*32 + ((n&15) | ((k&16)?16:0)))*16 + (k&15)
// so lane L of fragment `frag` reads 32 contiguous bytes.
__device__ inline int b_sw_index(int k, int n) {
  return ((((k >> 5) << 2) + (n >> 4)) << 9) +
         ((((n & 15) + ((k & 16) ? 16 : 0))) << 4) + (k & 15);
}
__device__ inline v16h load_b_sw(const _Float16* s, int frag, int lane) {
  const _Float16* p = s + (((frag << 5) + lane) << 4);
  v8h lo = *(const v8h*)(p);
  v8h hi = *(const v8h*)(p + 8);
  return __builtin_shufflevector(lo, hi, 0, 1, 2, 3, 4, 5, 6, 7,
                                 8, 9, 10, 11, 12, 13, 14, 15);
}

// ---------------------------------------------------------------------------
// Pass 1: measurement encoder + Q/K/V projection.  512 blocks x 256 threads.
// K/V go to global memory in per-64-row-tile swizzled B layout.
// ---------------------------------------------------------------------------
__global__ __launch_bounds__(256) void enc_kernel(
    const float* __restrict__ meas,
    const float* __restrict__ w1, const float* __restrict__ b1,
    const float* __restrict__ w2, const float* __restrict__ b2,
    const float* __restrict__ qw, const float* __restrict__ qb,
    const float* __restrict__ kw, const float* __restrict__ kbias,
    const float* __restrict__ vw, const float* __restrict__ vb,
    _Float16* __restrict__ q16, _Float16* __restrict__ k16,
    _Float16* __restrict__ v16, _Float16* __restrict__ feat16) {
  __shared__ alignas(16) _Float16 s_act[QT * NH];   // h1 (f16)         16 KB
  __shared__ alignas(16) _Float16 s_act2[QT * NH];  // feat (f16)       16 KB
  __shared__ alignas(16) _Float16 s_w[3][NH * NH];  // swizzled weights 24 KB

  const int tid  = threadIdx.x;
  const int lane = tid & 31;
  const int wave = tid >> 5;
  const int b    = blockIdx.x >> 4;
  const int tile = blockIdx.x & 15;
  const size_t base = (size_t)b * NM + (size_t)tile * QT;  // first row

  // h1 = relu(x @ W1 + b1), K=5: pure VALU, 32 outputs/thread.
#pragma unroll 4
  for (int i = 0; i < 32; ++i) {
    int o = tid + 256 * i;            // 8192 outputs
    int row = o >> 6, h = o & 63;
    const float* x = meas + (base + row) * 5;
    float s = b1[h];
#pragma unroll
    for (int k = 0; k < 5; ++k) s += x[k] * w1[k * NH + h];
    s_act[row * NH + h] = (_Float16)fmaxf(s, 0.0f);
  }
  // stage me_w2 (64x64, K-major) into swizzled fragment layout
  for (int i = 0; i < 16; ++i) {
    int o = tid + 256 * i;
    int k = o >> 6, n = o & 63;
    s_w[0][b_sw_index(k, n)] = (_Float16)w2[o];
  }
  __syncthreads();

  // feat = relu(h1 @ W2 + b2) via WMMA: wave owns 16 rows, 4 N-tiles, K=64.
  {
    int r0 = wave * 16;
    v16h A0 = make_a_frag(s_act, lane, r0, 0, NH);
    v16h A1 = make_a_frag(s_act, lane, r0, 32, NH);
#pragma unroll
    for (int n = 0; n < 4; ++n) {
      v8f c = {};
      v16h B0 = load_b_sw(s_w[0], n, lane);
      v16h B1 = load_b_sw(s_w[0], 4 + n, lane);
      c = __builtin_amdgcn_wmma_f32_16x16x32_f16(false, A0, false, B0,
                                                 (short)0, c, false, false);
      c = __builtin_amdgcn_wmma_f32_16x16x32_f16(false, A1, false, B1,
                                                 (short)0, c, false, false);
      int ncol  = 16 * n + (lane & 15);
      int rbase = r0 + ((lane < 16) ? 0 : 8);
      float bias = b2[ncol];
#pragma unroll
      for (int r = 0; r < 8; ++r)
        s_act2[(rbase + r) * NH + ncol] = (_Float16)fmaxf(c[r] + bias, 0.0f);
    }
  }
  __syncthreads();  // all waves done reading s_w[0] / writing s_act2

  // stage q/k/v weights (swizzled)
  for (int i = 0; i < 16; ++i) {
    int o = tid + 256 * i;
    int k = o >> 6, n = o & 63;
    int idx = b_sw_index(k, n);
    s_w[0][idx] = (_Float16)qw[o];
    s_w[1][idx] = (_Float16)kw[o];
    s_w[2][idx] = (_Float16)vw[o];
  }
  __syncthreads();

  {
    int r0 = wave * 16;
    v16h A0 = make_a_frag(s_act2, lane, r0, 0, NH);
    v16h A1 = make_a_frag(s_act2, lane, r0, 32, NH);
    const float* biasp[3] = {qb, kbias, vb};
#pragma unroll
    for (int w = 0; w < 3; ++w) {
      float scale = (w == 0) ? 0.125f : 1.0f;  // fold 1/sqrt(H) into Q
#pragma unroll
      for (int n = 0; n < 4; ++n) {
        v8f c = {};
        v16h B0 = load_b_sw(s_w[w], n, lane);
        v16h B1 = load_b_sw(s_w[w], 4 + n, lane);
        c = __builtin_amdgcn_wmma_f32_16x16x32_f16(false, A0, false, B0,
                                                   (short)0, c, false, false);
        c = __builtin_amdgcn_wmma_f32_16x16x32_f16(false, A1, false, B1,
                                                   (short)0, c, false, false);
        int ncol  = 16 * n + (lane & 15);
        int rbase = r0 + ((lane < 16) ? 0 : 8);
        float bb  = biasp[w][ncol];
#pragma unroll
        for (int r = 0; r < 8; ++r) {
          _Float16 val = (_Float16)((c[r] + bb) * scale);
          if (w == 0) {
            // Q stays row-major (consumed as WMMA A operand)
            q16[(base + rbase + r) * NH + ncol] = val;
          } else {
            // K/V written straight into per-tile swizzled B layout
            int rowb = tile * QT + rbase + r;     // row within batch
            int tau  = rowb >> 6, rloc = rowb & 63;
            size_t blockoff = ((size_t)(b * (NM / KVT) + tau)) * TILE_ELEMS;
            if (w == 1)
              k16[blockoff + b_sw_index(ncol, rloc)] = val;  // k-dim = feature
            else
              v16[blockoff + b_sw_index(rloc, ncol)] = val;  // k-dim = kv row
          }
        }
      }
    }
  }
  // residual copy of feat in f16 (vectorized, 16B chunks)
#pragma unroll
  for (int i = 0; i < 4; ++i) {
    int o = (tid + 256 * i) * 8;
    *(v8h*)(feat16 + base * NH + o) = *(const v8h*)(s_act2 + o);
  }
}

// ---------------------------------------------------------------------------
// Pass 2: flash attention + residual + fused a1 MLP.  512 blocks x 256 thr.
// ---------------------------------------------------------------------------
__global__ __launch_bounds__(256) void attn_kernel(
    const _Float16* __restrict__ q16, const _Float16* __restrict__ k16,
    const _Float16* __restrict__ v16, const _Float16* __restrict__ feat16,
    const float* __restrict__ a1w1, const float* __restrict__ a1b1,
    const float* __restrict__ a1w2, const float* __restrict__ a1b2,
    float* __restrict__ feat2, float* __restrict__ a1out) {
  __shared__ alignas(16) _Float16 s_q[QT * NH];     // Q / reused feat  16 KB
  __shared__ alignas(16) _Float16 s_kT[TILE_ELEMS]; // K^T swizzled B    8 KB
  __shared__ alignas(16) _Float16 s_v[TILE_ELEMS];  // V swizzled B      8 KB
  __shared__ alignas(16) _Float16 s_p[8][16 * NH];  // per-wave P       16 KB

  const int tid  = threadIdx.x;
  const int lane = tid & 31;
  const int wave = tid >> 5;
  const int b    = blockIdx.x >> 4;
  const int tile = blockIdx.x & 15;
  const size_t base = (size_t)b * NM + (size_t)tile * QT;
  const size_t qoff = base * NH;

  // stage Q tile (16B chunks)
#pragma unroll
  for (int i = 0; i < 4; ++i) {
    int o = (tid + 256 * i) * 8;
    *(v8h*)(s_q + o) = *(const v8h*)(q16 + qoff + o);
  }
  __syncthreads();

  const int r0 = wave * 16;
  v16h QA0 = make_a_frag(s_q, lane, r0, 0, NH);
  v16h QA1 = make_a_frag(s_q, lane, r0, 32, NH);

  v8f zf = {};
  v8f O[4];
  float mrow[8], lrow[8];
#pragma unroll
  for (int n = 0; n < 4; ++n) O[n] = zf;
#pragma unroll
  for (int r = 0; r < 8; ++r) { mrow[r] = -1e30f; lrow[r] = 0.0f; }

  for (int j = 0; j < NM / KVT; ++j) {
    // KV tiles are pre-swizzled in global memory: verbatim contiguous copy,
    // 32 B per thread per array, via async load-to-LDS when available.
    const _Float16* kt = k16 + ((size_t)(b * (NM / KVT) + j)) * TILE_ELEMS;
    const _Float16* vt = v16 + ((size_t)(b * (NM / KVT) + j)) * TILE_ELEMS;
    {
      int o = tid * 16;
      copy16_to_lds(kt + o, s_kT + o);
      copy16_to_lds(kt + o + 8, s_kT + o + 8);
      copy16_to_lds(vt + o, s_v + o);
      copy16_to_lds(vt + o + 8, s_v + o + 8);
    }
    if (j + 1 < NM / KVT) {             // hint the next tile toward cache
      __builtin_prefetch(kt + TILE_ELEMS, 0, 1);
      __builtin_prefetch(vt + TILE_ELEMS, 0, 1);
    }
    async_copy_wait();
    __syncthreads();

    // S = Q @ K^T (16 x 64 per wave), f32 accumulation.
    v8f S[4];
#pragma unroll
    for (int n = 0; n < 4; ++n) {
      S[n] = zf;
      v16h B0 = load_b_sw(s_kT, n, lane);
      v16h B1 = load_b_sw(s_kT, 4 + n, lane);
      S[n] = __builtin_amdgcn_wmma_f32_16x16x32_f16(false, QA0, false, B0,
                                                    (short)0, S[n], false, false);
      S[n] = __builtin_amdgcn_wmma_f32_16x16x32_f16(false, QA1, false, B1,
                                                    (short)0, S[n], false, false);
    }

    // online softmax (rows live in 16-lane halves; xor-reduce inside halves)
    const int rbase = (lane < 16) ? 0 : 8;
    const int ncol0 = lane & 15;
#pragma unroll
    for (int r = 0; r < 8; ++r) {
      float rmax = fmaxf(fmaxf(S[0][r], S[1][r]), fmaxf(S[2][r], S[3][r]));
#pragma unroll
      for (int off = 1; off < 16; off <<= 1)
        rmax = fmaxf(rmax, __shfl_xor(rmax, off));
      float mn = fmaxf(mrow[r], rmax);
      float sc = __expf(mrow[r] - mn);
      mrow[r] = mn;
      lrow[r] *= sc;
      O[0][r] *= sc; O[1][r] *= sc; O[2][r] *= sc; O[3][r] *= sc;
      float rsum = 0.0f;
#pragma unroll
      for (int n = 0; n < 4; ++n) {
        float p = __expf(S[n][r] - mn);
        S[n][r] = p;
        rsum += p;
      }
#pragma unroll
      for (int off = 1; off < 16; off <<= 1) rsum += __shfl_xor(rsum, off);
      lrow[r] += rsum;
#pragma unroll
      for (int n = 0; n < 4; ++n)
        s_p[wave][(rbase + r) * NH + 16 * n + ncol0] = (_Float16)S[n][r];
    }
    __syncthreads();  // s_p C->A re-layout visible wave-wide

    // O += P @ V
    v16h PA0 = make_a_frag(s_p[wave], lane, 0, 0, NH);
    v16h PA1 = make_a_frag(s_p[wave], lane, 0, 32, NH);
#pragma unroll
    for (int n = 0; n < 4; ++n) {
      v16h B0 = load_b_sw(s_v, n, lane);
      v16h B1 = load_b_sw(s_v, 4 + n, lane);
      O[n] = __builtin_amdgcn_wmma_f32_16x16x32_f16(false, PA0, false, B0,
                                                    (short)0, O[n], false, false);
      O[n] = __builtin_amdgcn_wmma_f32_16x16x32_f16(false, PA1, false, B1,
                                                    (short)0, O[n], false, false);
    }
    __syncthreads();  // safe to overwrite s_kT/s_v/s_p next iteration
  }

  // normalize, add residual, store feat (f32 + f16 into reused s_q)
  {
    const int rbase = (lane < 16) ? 0 : 8;
#pragma unroll
    for (int n = 0; n < 4; ++n) {
      int ncol = 16 * n + (lane & 15);
#pragma unroll
      for (int r = 0; r < 8; ++r) {
        int row   = r0 + rbase + r;
        float val = O[n][r] / lrow[r] +
                    (float)feat16[(base + row) * NH + ncol];
        feat2[(base + row) * NH + ncol] = val;
        s_q[row * NH + ncol] = (_Float16)val;
      }
    }
  }
  __syncthreads();

  // fused a1 = relu(feat @ a1_w1 + b1) @ a1_w2 + b2, one row per thread
  if (tid < QT) {
    int row = tid;
    float acc = a1b2[0];
#pragma unroll 4
    for (int h2 = 0; h2 < 32; ++h2) {
      float s = a1b1[h2];
      for (int k = 0; k < NH; ++k)
        s += (float)s_q[row * NH + k] * a1w1[k * 32 + h2];
      acc += fmaxf(s, 0.0f) * a1w2[h2];
    }
    a1out[base + row] = acc;
  }
}

// ---------------------------------------------------------------------------
// Pass 3: segment softmax over tags, tag aggregation, heads. 32 blocks.
// ---------------------------------------------------------------------------
__global__ __launch_bounds__(256) void agg_kernel(
    const float* __restrict__ feat2, const float* __restrict__ a1,
    const int* __restrict__ map,
    const float* __restrict__ a2w1, const float* __restrict__ a2b1,
    const float* __restrict__ a2w2, const float* __restrict__ a2b2,
    const float* __restrict__ muw, const float* __restrict__ mub,
    const float* __restrict__ lw, const float* __restrict__ lb,
    float* __restrict__ out) {
  __shared__ float s_red[256 * NT];      // 8 KB
  __shared__ float s_max[NT];
  __shared__ float s_tf[4][NT][NH];      // 8 KB partial tag feats
  __shared__ float s_se[4][NT];
  __shared__ float s_set[NT];
  __shared__ float s_tfn[NT][NH];        // normalized tag feats
  __shared__ float s_logit[NT];
  __shared__ float s_wgt[NT];
  __shared__ float s_pose[NH];

  const int b   = blockIdx.x;
  const int tid = threadIdx.x;
  const float* a1b = a1 + (size_t)b * NM;
  const int*   mp  = map + (size_t)b * NM;
  const float* fb  = feat2 + (size_t)b * NM * NH;

  // per-tag max of a1
  float tmax[NT];
#pragma unroll
  for (int t = 0; t < NT; ++t) tmax[t] = -1e30f;
  for (int m = tid; m < NM; m += 256) {
    int t = mp[m];
    tmax[t] = fmaxf(tmax[t], a1b[m]);
  }
#pragma unroll
  for (int t = 0; t < NT; ++t) s_red[tid * NT + t] = tmax[t];
  __syncthreads();
  if (tid < NT) {
    float mx = -1e30f;
    for (int i = 0; i < 256; ++i) mx = fmaxf(mx, s_red[i * NT + tid]);
    s_max[tid] = mx;
  }
  __syncthreads();

  // segment-softmax weighted tag features
  const int g = tid >> 6, h = tid & 63;
  float acc[NT], se[NT];
#pragma unroll
  for (int t = 0; t < NT; ++t) { acc[t] = 0.0f; se[t] = 0.0f; }
  for (int m = g; m < NM; m += 4) {
    int t = mp[m];
    float e = __expf(a1b[m] - s_max[t]);
    acc[t] += e * fb[(size_t)m * NH + h];
    se[t]  += e;
  }
#pragma unroll
  for (int t = 0; t < NT; ++t) s_tf[g][t][h] = acc[t];
  if (h == 0)
#pragma unroll
    for (int t = 0; t < NT; ++t) s_se[g][t] = se[t];
  __syncthreads();

  for (int idx = tid; idx < NT * NH; idx += 256) {
    int t = idx >> 6, hh = idx & 63;
    s_tfn[t][hh] = s_tf[0][t][hh] + s_tf[1][t][hh] + s_tf[2][t][hh] +
                   s_tf[3][t][hh];
  }
  if (tid < NT)
    s_set[tid] = s_se[0][tid] + s_se[1][tid] + s_se[2][tid] + s_se[3][tid];
  __syncthreads();
  for (int idx = tid; idx < NT * NH; idx += 256) {
    int t = idx >> 6, hh = idx & 63;
    s_tfn[t][hh] /= s_set[t];
  }
  __syncthreads();

  // a2 MLP per tag
  if (tid < NT) {
    int t = tid;
    float aacc = a2b2[0];
    for (int h2 = 0; h2 < 32; ++h2) {
      float s = a2b1[h2];
      for (int k = 0; k < NH; ++k) s += s_tfn[t][k] * a2w1[k * 32 + h2];
      aacc += fmaxf(s, 0.0f) * a2w2[h2];
    }
    s_logit[t] = aacc;
  }
  __syncthreads();
  if (tid == 0) {
    float mx = -1e30f;
    for (int t = 0; t < NT; ++t) mx = fmaxf(mx, s_logit[t]);
    float sum = 0.0f;
    for (int t = 0; t < NT; ++t) {
      float e = __expf(s_logit[t] - mx);
      s_wgt[t] = e;
      sum += e;
    }
    for (int t = 0; t < NT; ++t) s_wgt[t] /= sum;
  }
  __syncthreads();
  if (tid < NH) {
    float p = 0.0f;
    for (int t = 0; t < NT; ++t) p += s_wgt[t] * s_tfn[t][tid];
    s_pose[tid] = p;
  }
  __syncthreads();
  if (tid < 3) {  // mu
    float s = mub[tid];
    for (int k = 0; k < NH; ++k) s += s_pose[k] * muw[k * 3 + tid];
    out[b * 3 + tid] = s;
  } else if (tid >= 32 && tid < 38) {  // logL
    int j = tid - 32;
    float s = lb[j];
    for (int k = 0; k < NH; ++k) s += s_pose[k] * lw[k * 6 + j];
    out[NB * 3 + b * 6 + j] = s;
  }
}

// ---------------------------------------------------------------------------
extern "C" void kernel_launch(void* const* d_in, const int* in_sizes, int n_in,
                              void* d_out, int out_size, void* d_ws,
                              size_t ws_size, hipStream_t stream) {
  (void)in_sizes; (void)n_in; (void)out_size; (void)ws_size;
  const float* meas = (const float*)d_in[0];
  const int*   map  = (const int*)d_in[1];
  const float* me_w1 = (const float*)d_in[2];
  const float* me_b1 = (const float*)d_in[3];
  const float* me_w2 = (const float*)d_in[4];
  const float* me_b2 = (const float*)d_in[5];
  const float* q_w = (const float*)d_in[6];
  const float* q_b = (const float*)d_in[7];
  const float* k_w = (const float*)d_in[8];
  const float* k_b = (const float*)d_in[9];
  const float* v_w = (const float*)d_in[10];
  const float* v_b = (const float*)d_in[11];
  const float* a1_w1 = (const float*)d_in[12];
  const float* a1_b1 = (const float*)d_in[13];
  const float* a1_w2 = (const float*)d_in[14];
  const float* a1_b2 = (const float*)d_in[15];
  const float* a2_w1 = (const float*)d_in[16];
  const float* a2_b1 = (const float*)d_in[17];
  const float* a2_w2 = (const float*)d_in[18];
  const float* a2_b2 = (const float*)d_in[19];
  const float* mu_w  = (const float*)d_in[20];
  const float* mu_b  = (const float*)d_in[21];
  const float* lo_w  = (const float*)d_in[22];
  const float* lo_b  = (const float*)d_in[23];

  const size_t n16 = (size_t)NB * NM * NH;  // 4,194,304 elements
  _Float16* q16    = (_Float16*)d_ws;
  _Float16* k16    = q16 + n16;
  _Float16* v16    = k16 + n16;
  _Float16* feat16 = v16 + n16;
  float*    feat2  = (float*)(feat16 + n16);
  float*    a1buf  = feat2 + n16;

  dim3 blk(256);
  enc_kernel<<<dim3(NB * (NM / QT)), blk, 0, stream>>>(
      meas, me_w1, me_b1, me_w2, me_b2, q_w, q_b, k_w, k_b, v_w, v_b,
      q16, k16, v16, feat16);
  attn_kernel<<<dim3(NB * (NM / QT)), blk, 0, stream>>>(
      q16, k16, v16, feat16, a1_w1, a1_b1, a1_w2, a1_b2, feat2, a1buf);
  agg_kernel<<<dim3(NB), blk, 0, stream>>>(
      feat2, a1buf, map, a2_w1, a2_b1, a2_w2, a2_b2, mu_w, mu_b, lo_w, lo_b,
      (float*)d_out);
}
